// generate_graph_23673859735697
// MI455X (gfx1250) — matrix operand order repaired and checked by reading, not compiled, not run on hardware
//
#include <hip/hip_runtime.h>
#include <hip/hip_bf16.h>
#include <math.h>

// ---------------------------------------------------------------------------
// Problem constants
// ---------------------------------------------------------------------------
#define NNODES 10000
#define CIN    128
#define DOUT   20
#define DIMP   32          // padded feature dim for WMMA (K = 32)
#define KNN    16
#define NTILES 625         // 10000 / 16
#define NPAIRS 313         // ceil(625 / 2)
#define NITERS 79          // ceil(313 / 4)  (4 waves per workgroup)
#define BN_EPS 1e-5f

typedef __attribute__((ext_vector_type(16))) _Float16 v16h;
typedef __attribute__((ext_vector_type(8)))  _Float16 v8h;
typedef __attribute__((ext_vector_type(8)))  float    v8f;

// Workspace layout (bytes, 256-aligned blocks)
#define OFF_H     0u         //  h          : N*20 f32  (800000 B)
#define OFF_STATS 800256u    //  mu/invstd  : 40 f32
#define OFF_SQ    800512u    //  |emb|^2    : N f32     (40000 B)
#define OFF_EMBH  840704u    //  emb f16 pad: N*32 f16  (640000 B)
#define OFF_EMBP  1480704u   //  emb+noise  : N*20 f32  (800000 B)
#define OFF_KE    2280704u   //  knn emb idx: N*16 i32  (640000 B)
#define OFF_KP    2920704u   //  knn pos idx: N*16 i32  (640000 B)

// Output layout (floats)
#define O_P     0u           // p                      [160000]
#define O_SIV_P 160000u      // soft_index_v row 0 = p [160000]
#define O_SIV_D 320000u      // soft_index_v row 1 = dst
#define O_EL_S  480000u      // edges_large row 0 (src)
#define O_EL_D  640000u      // edges_large row 1 (dst)
#define O_EI_S0 800000u      // edge_index row 0: emb src
#define O_EI_S1 960000u      // edge_index row 0: pos src
#define O_EI_D0 1120000u     // edge_index row 1: emb dst
#define O_EI_D1 1280000u     // edge_index row 1: pos dst

// ---------------------------------------------------------------------------
// Register-resident streaming top-16 (tracked-worst replacement)
// ---------------------------------------------------------------------------
__device__ __forceinline__ void topk_insert(float (&bd)[KNN], int (&bi)[KNN],
                                            float& wv, int& wslot,
                                            float d, int c) {
  if (d < wv) {
#pragma unroll
    for (int q = 0; q < KNN; ++q)
      if (q == wslot) { bd[q] = d; bi[q] = c; }
    float nv = bd[0]; int ns = 0;
#pragma unroll
    for (int q = 1; q < KNN; ++q)
      if (bd[q] > nv) { nv = bd[q]; ns = q; }
    wv = nv; wslot = ns;
  }
}

// ---------------------------------------------------------------------------
// 1) h = x @ W + b        (51 MFLOP, VALU is fine; W stays hot in cache)
// ---------------------------------------------------------------------------
__global__ __launch_bounds__(256) void gg_mlp_kernel(
    const float* __restrict__ x, const float* __restrict__ W,
    const float* __restrict__ b, float* __restrict__ h) {
  int e = blockIdx.x * 256 + threadIdx.x;
  if (e >= NNODES * DOUT) return;
  int i = e / DOUT, d = e % DOUT;
  const float* xr = x + (size_t)i * CIN;
  float acc = b[d];
#pragma unroll 8
  for (int c = 0; c < CIN; ++c) acc = fmaf(xr[c], W[c * DOUT + d], acc);
  h[e] = acc;
}

// ---------------------------------------------------------------------------
// 2) BatchNorm training stats: mu[d], invstd[d]   (single workgroup)
// ---------------------------------------------------------------------------
__global__ __launch_bounds__(256) void gg_bnstats_kernel(
    const float* __restrict__ h, float* __restrict__ stats) {
  __shared__ float s1[256][DOUT];
  __shared__ float s2[256][DOUT];
  int t = threadIdx.x;
  float a1[DOUT], a2[DOUT];
#pragma unroll
  for (int d = 0; d < DOUT; ++d) { a1[d] = 0.f; a2[d] = 0.f; }
  for (int i = t; i < NNODES; i += 256) {
    const float* hr = h + (size_t)i * DOUT;
#pragma unroll
    for (int d = 0; d < DOUT; ++d) {
      float v = hr[d];
      a1[d] += v; a2[d] += v * v;
    }
  }
#pragma unroll
  for (int d = 0; d < DOUT; ++d) { s1[t][d] = a1[d]; s2[t][d] = a2[d]; }
  __syncthreads();
  for (int off = 128; off > 0; off >>= 1) {
    if (t < off) {
#pragma unroll
      for (int d = 0; d < DOUT; ++d) {
        s1[t][d] += s1[t + off][d];
        s2[t][d] += s2[t + off][d];
      }
    }
    __syncthreads();
  }
  if (t < DOUT) {
    float mu  = s1[0][t] * (1.0f / NNODES);
    float var = s2[0][t] * (1.0f / NNODES) - mu * mu;
    stats[t]        = mu;
    stats[DOUT + t] = rsqrtf(var + BN_EPS);
  }
}

// ---------------------------------------------------------------------------
// 3) emb = relu(BN(h));  pack f16 (padded to 32), row norms, perturbed copy
// ---------------------------------------------------------------------------
__global__ __launch_bounds__(256) void gg_bnpack_kernel(
    const float* __restrict__ h, const float* __restrict__ noise,
    const float* __restrict__ gamma, const float* __restrict__ beta,
    const float* __restrict__ stats,
    _Float16* __restrict__ embh, float* __restrict__ sq,
    float* __restrict__ embp) {
  int i = blockIdx.x * 256 + threadIdx.x;
  if (i >= NNODES) return;
  const float* hr = h + (size_t)i * DOUT;
  const float* nr = noise + (size_t)i * DOUT;
  _Float16* er = embh + (size_t)i * DIMP;
  float* pr = embp + (size_t)i * DOUT;
  float ssq = 0.f;
#pragma unroll
  for (int d = 0; d < DOUT; ++d) {
    float v = (hr[d] - stats[d]) * stats[DOUT + d] * gamma[d] + beta[d];
    v = fmaxf(v, 0.f);
    er[d] = (_Float16)v;
    ssq += v * v;
    pr[d] = v + nr[d] * 1e-4f;
  }
#pragma unroll
  for (int d = DOUT; d < DIMP; ++d) er[d] = (_Float16)0.f;
  sq[i] = ssq;
}

// ---------------------------------------------------------------------------
// 4) WMMA KNN on embeddings.
//    Block = 4 waves, owns one 16-row query tile.  Per iteration each wave
//    computes two 16x16 d^2 tiles via v_wmma_f32_16x16x32_f16
//    (d^2 = |q|^2 + |c|^2 - 2 q.c), transposes through LDS, and each lane
//    streams 16 candidates into a register top-16.  Final LDS merge of the
//    8 partial lists per row emits sorted 16-NN indices.
// ---------------------------------------------------------------------------
__global__ __launch_bounds__(128) void gg_knn_emb_kernel(
    const _Float16* __restrict__ embh, const float* __restrict__ sq,
    int* __restrict__ out_idx) {
  __shared__ float lds_d2[4][16][33];   // per-wave 16 rows x 32 cols (+pad)
  __shared__ float lds_bd[16][128];
  __shared__ int   lds_bi[16][128];

  const int wave = threadIdx.x >> 5;
  const int lane = threadIdx.x & 31;
  const int half = lane >> 4;
  const int l16  = lane & 15;
  const int qbase = blockIdx.x * 16;
  const float INF = __builtin_inff();

  // --- A tile: 16 queries x 32 (f16).  WMMA A layout:
  // lane(0-15):  M=lane, K=[half*8 .. +7] in elems 0-7, K=[16+half*8 ..] in 8-15
  const v8h* ar = (const v8h*)(embh + (size_t)(qbase + l16) * DIMP);
  v8h alo = ar[half];          // halves [half*8 .. half*8+7]
  v8h ahi = ar[2 + half];      // halves [16+half*8 .. ]
  v16h A;
#pragma unroll
  for (int e = 0; e < 8; ++e) { A[e] = alo[e]; A[8 + e] = ahi[e]; }

  // query norms for the rows this lane's accumulator covers (M = v + 8*half)
  float qs[8];
#pragma unroll
  for (int v = 0; v < 8; ++v) qs[v] = sq[qbase + 8 * half + v];

  float bestd[KNN]; int besti[KNN];
#pragma unroll
  for (int q = 0; q < KNN; ++q) { bestd[q] = INF; besti[q] = 0; }
  float wv = INF; int wslot = 0;

  for (int ii = 0; ii < NITERS; ++ii) {
    const int tp = ii * 4 + wave;           // candidate tile-pair index
    if (tp < NPAIRS) {
      if (tp + 4 < NPAIRS)
        __builtin_prefetch(embh + (size_t)((tp + 4) * 32) * DIMP, 0, 1);
#pragma unroll
      for (int sub = 0; sub < 2; ++sub) {
        const int tile = tp * 2 + sub;       // wave-uniform
        if (tile < NTILES) {
          const int c = tile * 16 + l16;
          // B tile (32x16, K=feature, N=candidate):
          // lane(0-15): N=lane, K=[half*8 .. +7] in elems 0-7, K=[16+half*8 ..] in 8-15
          const v8h* br = (const v8h*)(embh + (size_t)c * DIMP);
          v8h blo = br[half];
          v8h bhi = br[2 + half];
          v16h B;
#pragma unroll
          for (int e = 0; e < 8; ++e) { B[e] = blo[e]; B[8 + e] = bhi[e]; }
          v8f acc = {};
          acc = __builtin_amdgcn_wmma_f32_16x16x32_f16(
              false, A, false, B, (short)0, acc, false, false);
          const float cs = sq[c];
#pragma unroll
          for (int v = 0; v < 8; ++v)
            lds_d2[wave][v + 8 * half][sub * 16 + l16] =
                qs[v] + cs - 2.0f * acc[v];
        } else {
#pragma unroll
          for (int v = 0; v < 8; ++v)
            lds_d2[wave][v + 8 * half][sub * 16 + l16] = INF;
        }
      }
    }
    __syncthreads();
    if (tp < NPAIRS) {
      const int scanTile = tp * 2 + half;    // lane-half picks sub-tile
      if (scanTile < NTILES) {
        const int cb = scanTile * 16;
        const int qi = qbase + l16;          // this lane's query row
        for (int k = 0; k < 16; ++k) {
          const int c = cb + k;
          if (c == qi) continue;             // no self loops
          topk_insert(bestd, besti, wv, wslot,
                      lds_d2[wave][l16][half * 16 + k], c);
        }
      }
    }
    __syncthreads();
  }

  // --- merge 8 partial lists per row, emit sorted ascending (top_k order) ---
  const int slot = (wave * 2 + half) * 16;
#pragma unroll
  for (int q = 0; q < KNN; ++q) {
    lds_bd[l16][slot + q] = bestd[q];
    lds_bi[l16][slot + q] = besti[q];
  }
  __syncthreads();
  if (threadIdx.x < 16) {
    const int row = threadIdx.x;
    for (int k = 0; k < KNN; ++k) {
      float mv = INF; int mi = 0, ms = -1;
      for (int q = 0; q < 128; ++q) {
        float v = lds_bd[row][q];
        if (v < mv) { mv = v; mi = lds_bi[row][q]; ms = q; }
      }
      if (ms >= 0) lds_bd[row][ms] = INF;
      out_idx[(size_t)(qbase + row) * KNN + k] = mi;
    }
  }
}

// ---------------------------------------------------------------------------
// 5) KNN on positions (3-D): 4 segment-threads per row + LDS merge
// ---------------------------------------------------------------------------
__global__ __launch_bounds__(256) void gg_knn_pos_kernel(
    const float* __restrict__ pos, int* __restrict__ out_idx) {
  __shared__ float sd[256][KNN];
  __shared__ int   si[256][KNN];
  const int gtid = blockIdx.x * 256 + threadIdx.x;
  const int row = gtid >> 2;
  const int seg = gtid & 3;
  const bool active = row < NNODES;
  const float INF = __builtin_inff();

  float bestd[KNN]; int besti[KNN];
#pragma unroll
  for (int q = 0; q < KNN; ++q) { bestd[q] = INF; besti[q] = 0; }
  float wv = INF; int wslot = 0;

  if (active) {
    const float px = pos[row * 3 + 0];
    const float py = pos[row * 3 + 1];
    const float pz = pos[row * 3 + 2];
    const int j0 = seg * (NNODES / 4), j1 = j0 + (NNODES / 4);
    for (int j = j0; j < j1; ++j) {
      if (j == row) continue;
      float dx = pos[j * 3 + 0] - px;
      float dy = pos[j * 3 + 1] - py;
      float dz = pos[j * 3 + 2] - pz;
      topk_insert(bestd, besti, wv, wslot, dx * dx + dy * dy + dz * dz, j);
    }
  }
#pragma unroll
  for (int q = 0; q < KNN; ++q) {
    sd[threadIdx.x][q] = bestd[q];
    si[threadIdx.x][q] = besti[q];
  }
  __syncthreads();
  if (active && seg == 0) {
    const int b = threadIdx.x;   // segments at b .. b+3
    for (int k = 0; k < KNN; ++k) {
      float mv = INF; int mi = 0, ms = -1, mq = 0;
      for (int s = 0; s < 4; ++s)
        for (int q = 0; q < KNN; ++q) {
          float v = sd[b + s][q];
          if (v < mv) { mv = v; mi = si[b + s][q]; ms = b + s; mq = q; }
        }
      if (ms >= 0) sd[ms][mq] = INF;
      out_idx[(size_t)row * KNN + k] = mi;
    }
  }
}

// ---------------------------------------------------------------------------
// 6) Edge distances on perturbed emb, p = exp(-t*dist), scatter all outputs
// ---------------------------------------------------------------------------
__global__ __launch_bounds__(256) void gg_edges_kernel(
    const int* __restrict__ knn_e, const int* __restrict__ knn_p,
    const float* __restrict__ embp, const float* __restrict__ t,
    float* __restrict__ out) {
  int e = blockIdx.x * 256 + threadIdx.x;
  if (e >= NNODES * KNN) return;
  const int i = e >> 4;
  const int js = knn_e[e];
  const float* a = embp + (size_t)js * DOUT;
  const float* b = embp + (size_t)i * DOUT;
  float ss = 0.f;
#pragma unroll
  for (int d = 0; d < DOUT; ++d) {
    float df = a[d] - b[d];
    ss = fmaf(df, df, ss);
  }
  const float p = expf(-t[0] * sqrtf(ss));
  const float fi = (float)i;
  out[O_P + e]     = p;
  out[O_SIV_P + e] = p;
  out[O_SIV_D + e] = fi;
  out[O_EL_S + e]  = (float)js;
  out[O_EL_D + e]  = fi;
  out[O_EI_S0 + e] = (float)js;
  out[O_EI_S1 + e] = (float)knn_p[e];
  out[O_EI_D0 + e] = fi;
  out[O_EI_D1 + e] = fi;
}

// ---------------------------------------------------------------------------
// Launch
// ---------------------------------------------------------------------------
extern "C" void kernel_launch(void* const* d_in, const int* in_sizes, int n_in,
                              void* d_out, int out_size, void* d_ws, size_t ws_size,
                              hipStream_t stream) {
  (void)in_sizes; (void)n_in; (void)out_size; (void)ws_size;
  const float* x     = (const float*)d_in[0];
  const float* pos   = (const float*)d_in[1];
  const float* noise = (const float*)d_in[2];
  const float* W     = (const float*)d_in[3];
  const float* b     = (const float*)d_in[4];
  const float* gamma = (const float*)d_in[5];
  const float* beta  = (const float*)d_in[6];
  const float* t     = (const float*)d_in[7];
  float* out = (float*)d_out;

  char* ws = (char*)d_ws;
  float*    h     = (float*)(ws + OFF_H);
  float*    stats = (float*)(ws + OFF_STATS);
  float*    sq    = (float*)(ws + OFF_SQ);
  _Float16* embh  = (_Float16*)(ws + OFF_EMBH);
  float*    embp  = (float*)(ws + OFF_EMBP);
  int*      knn_e = (int*)(ws + OFF_KE);
  int*      knn_p = (int*)(ws + OFF_KP);

  gg_mlp_kernel<<<(NNODES * DOUT + 255) / 256, 256, 0, stream>>>(x, W, b, h);
  gg_bnstats_kernel<<<1, 256, 0, stream>>>(h, stats);
  gg_bnpack_kernel<<<(NNODES + 255) / 256, 256, 0, stream>>>(
      h, noise, gamma, beta, stats, embh, sq, embp);
  gg_knn_emb_kernel<<<NTILES, 128, 0, stream>>>(embh, sq, knn_e);
  gg_knn_pos_kernel<<<(NNODES * 4 + 255) / 256, 256, 0, stream>>>(pos, knn_p);
  gg_edges_kernel<<<(NNODES * KNN + 255) / 256, 256, 0, stream>>>(
      knn_e, knn_p, embp, t, out);
}
